// Aggregators_87170656239792
// MI455X (gfx1250) — matrix operand order repaired and checked by reading, not compiled, not run on hardware
//
#include <hip/hip_runtime.h>

// ---------------- CDNA5 vector types ----------------
typedef __attribute__((ext_vector_type(16))) __bf16      v16bf;
typedef __attribute__((ext_vector_type(8)))  float       v8f;
typedef __attribute__((ext_vector_type(4)))  float       f4;
typedef __attribute__((ext_vector_type(4)))  unsigned    v4u;
typedef __attribute__((ext_vector_type(8)))  int         v8i;
typedef __attribute__((ext_vector_type(4)))  int         v4i;

// ---------------- problem shape ----------------
constexpr int kB = 8;     // graphs
constexpr int kN = 2048;  // nodes
constexpr int kD = 128;   // emb size

// ---------------- tiling ----------------
constexpr int BM = 64;            // M rows per workgroup
constexpr int BK = 32;            // K per step (= bf16 WMMA K)
constexpr int KT = kN / BK;       // 64 K-steps
constexpr int A32_STRIDE = 36;    // fp32 LDS row stride (32 + 4 TDM pad dwords)
constexpr int BF_STRIDE  = 40;    // bf16 LDS row stride (keeps 16B alignment, kills conflicts)

struct __align__(16) Smem {
  float  aT32[2][BM][A32_STRIDE];   // TDM double-buffered fp32 A tiles (18,432 B)
  __bf16 aHi[BM][BF_STRIDE];        // split A, hi part   (5,120 B)
  __bf16 aLo[BM][BF_STRIDE];        // split A, lo part   (5,120 B)
  __bf16 bHi[kD][BF_STRIDE];        // split B^T, hi part (10,240 B)
  __bf16 bLo[kD][BF_STRIDE];        // split B^T, lo part (10,240 B)
};                                  // total 49,152 B

// ---------------- helpers ----------------
__device__ __forceinline__ unsigned lds_off(const void* p) {
  // low 32 bits of a generic pointer into the LDS aperture == LDS byte offset
  return (unsigned)(unsigned long long)(uintptr_t)p;
}

__device__ __forceinline__ void split_bf16(float x, __bf16& hi, __bf16& lo) {
  __bf16 h = (__bf16)x;
  hi = h;
  lo = (__bf16)(x - (float)h);
}

__device__ __forceinline__ v16bf ld_frag_a(const __bf16* row, int khalf) {
  // A 16x32 bf16 layout (ISA 7.12.2): lanes<16 hold K {0..7, 16..23},
  // lanes>=16 hold K {8..15, 24..31}; elems 0..7 = first run, 8..15 = +16 run.
  union { v16bf v; f4 q[2]; } u;
  u.q[0] = *reinterpret_cast<const f4*>(row + khalf * 8);
  u.q[1] = *reinterpret_cast<const f4*>(row + 16 + khalf * 8);
  return u.v;
}

__device__ __forceinline__ v16bf ld_frag_b(const __bf16* col, int khalf) {
  // B 32x16 bf16: lane n holds column n; lanes<16 K {0..7, 16..23},
  // lanes>=16 K {8..15, 24..31} (same K order as A).
  union { v16bf v; f4 q[2]; } u;
  const __bf16* p = col + khalf * 8;
  u.q[0] = *reinterpret_cast<const f4*>(p);
  u.q[1] = *reinterpret_cast<const f4*>(p + 16);
  return u.v;
}

// Issue one TDM copy: 64x32 fp32 tile of neibors -> LDS (with 4-dword row pad).
__device__ __forceinline__ void tdm_load_a(const float* gsrc, unsigned lds_byte_off) {
  unsigned long long ga = (unsigned long long)(uintptr_t)gsrc;
  v4u g0;
  g0[0] = 1u;                                              // count=1, user mode
  g0[1] = lds_byte_off;                                    // lds_addr
  g0[2] = (unsigned)(ga & 0xFFFFFFFFu);                    // global_addr[31:0]
  g0[3] = (unsigned)((ga >> 32) & 0x1FFFFFFu) | (2u << 30);// addr[56:32] | type=2
  v8i g1;
  g1[0] = (int)((2u << 16)     // data_size = 4B
              | (1u << 20)     // pad_enable
              | (4u << 22)     // pad_interval: every 32 dwords
              | (3u << 25));   // pad_amount: 4 dwords -> row stride 36 dwords
  g1[1] = (int)((unsigned)(kN & 0xFFFF) << 16);  // tensor_dim0 = 2048
  g1[2] = (int)((unsigned)(kN & 0xFFFF) << 16);  // tensor_dim1 = 2048
  g1[3] = (int)((unsigned)BK << 16);             // tile_dim0 = 32
  g1[4] = BM;                                    // tile_dim1 = 64
  g1[5] = kN;                                    // tensor_dim0_stride = 2048
  g1[6] = 0;
  g1[7] = 0;
  v4i z4 = {0, 0, 0, 0};
  v8i z8 = {0, 0, 0, 0, 0, 0, 0, 0};
  // amdgpu-toolchain (clang-23) 6-arg form: (g0, g1, g2, g3, g4, cpol)
  __builtin_amdgcn_tensor_load_to_lds(g0, g1, z4, z4, z8, 0);
}

// ---------------- kernel ----------------
__global__ __launch_bounds__(256) void Aggregators_87170656239792_kernel(
    const float* __restrict__ embs,   // [B][N][D]
    const float* __restrict__ neib,   // [B][N][N]
    float* __restrict__ out) {        // [B][N][D]
  __shared__ Smem sm;

  const int b    = blockIdx.y;
  const int m0   = blockIdx.x * BM;
  const int tid  = threadIdx.x;
  const int lane = tid & 31;
  const int wid  = tid >> 5;

  // wave-level fragment coordinates
  const int fm    = lane & 15;        // M row / B column within 16
  const int khalf = lane >> 4;        // which K half this lane holds
  const int dcol  = wid * 16 + fm;    // this lane's output column (0..127)

  // B-staging coordinates: thread loads one 16-wide chunk of one k-row
  const int skk = tid >> 3;           // 0..31 : k within tile
  const int sdb = (tid & 7) * 16;     // d block start

  // A-convert coordinates: thread converts 8 contiguous fp32 of one row
  const int car = tid >> 2;           // 0..63 : row
  const int cac = (tid & 3) * 8;      // col start

  const float* neib_b = neib + (size_t)b * kN * kN + (size_t)m0 * kN;
  const float* embs_b = embs + (size_t)b * kN * kD;

  v8f acc[4];
#pragma unroll
  for (int i = 0; i < 4; ++i)
    acc[i] = (v8f){0.f, 0.f, 0.f, 0.f, 0.f, 0.f, 0.f, 0.f};

  float bregs[16];

  // -------- prologue: TDM A(0) -> buf0, global-load B(0) into regs --------
  if (wid == 0) tdm_load_a(neib_b, lds_off(&sm.aT32[0][0][0]));
  {
    const float* gb = embs_b + (size_t)skk * kD + sdb;
#pragma unroll
    for (int j = 0; j < 16; j += 4) {
      f4 v = *reinterpret_cast<const f4*>(gb + j);
      bregs[j + 0] = v.x; bregs[j + 1] = v.y; bregs[j + 2] = v.z; bregs[j + 3] = v.w;
    }
  }
  if (wid == 0) __builtin_amdgcn_s_wait_tensorcnt(0);
  __syncthreads();

  // -------- main K loop --------
  for (int kt = 0; kt < KT; ++kt) {
    const int cur = kt & 1;
    const int nxt = cur ^ 1;

    // phase 1: split-convert A[cur] fp32 -> bf16 hi/lo; stage B regs -> LDS^T
    {
      const float* src = &sm.aT32[cur][car][cac];
#pragma unroll
      for (int j = 0; j < 8; ++j) {
        __bf16 h, l;
        split_bf16(src[j], h, l);
        sm.aHi[car][cac + j] = h;
        sm.aLo[car][cac + j] = l;
      }
#pragma unroll
      for (int j = 0; j < 16; ++j) {
        __bf16 h, l;
        split_bf16(bregs[j], h, l);
        sm.bHi[sdb + j][skk] = h;   // transposed store: column-major for B frags
        sm.bLo[sdb + j][skk] = l;
      }
    }

    // phase 2: prefetch next tile (TDM for A, global->regs for B)
    if (kt + 1 < KT) {
      if (wid == 0)
        tdm_load_a(neib_b + (size_t)(kt + 1) * BK, lds_off(&sm.aT32[nxt][0][0]));
      const float* gb = embs_b + (size_t)(kt + 1) * BK * kD + (size_t)skk * kD + sdb;
#pragma unroll
      for (int j = 0; j < 16; j += 4) {
        f4 v = *reinterpret_cast<const f4*>(gb + j);
        bregs[j + 0] = v.x; bregs[j + 1] = v.y; bregs[j + 2] = v.z; bregs[j + 3] = v.w;
      }
    }
    __syncthreads();

    // phase 3: 12 WMMAs (4 M-subtiles x 3 split terms)
    {
      v16bf bh = ld_frag_b(&sm.bHi[dcol][0], khalf);
      v16bf bl = ld_frag_b(&sm.bLo[dcol][0], khalf);
#pragma unroll
      for (int ms = 0; ms < 4; ++ms) {
        const int r = ms * 16 + fm;
        v16bf ah = ld_frag_a(&sm.aHi[r][0], khalf);
        v16bf al = ld_frag_a(&sm.aLo[r][0], khalf);
        acc[ms] = __builtin_amdgcn_wmma_f32_16x16x32_bf16(
            false, ah, false, bh, (short)0, acc[ms], false, false);
        acc[ms] = __builtin_amdgcn_wmma_f32_16x16x32_bf16(
            false, al, false, bh, (short)0, acc[ms], false, false);
        acc[ms] = __builtin_amdgcn_wmma_f32_16x16x32_bf16(
            false, ah, false, bl, (short)0, acc[ms], false, false);
      }
    }
    __syncthreads();
    if (wid == 0) __builtin_amdgcn_s_wait_tensorcnt(0);
    __syncthreads();
  }

  // -------- epilogue: C/D layout (ISA 7.12.2): VGPR j -> M = j + 8*khalf --------
  float* outp = out + (size_t)b * kN * kD;
#pragma unroll
  for (int ms = 0; ms < 4; ++ms) {
#pragma unroll
    for (int j = 0; j < 8; ++j) {
      const int row = m0 + ms * 16 + khalf * 8 + j;
      outp[(size_t)row * kD + dcol] = acc[ms][j];
    }
  }
}

// ---------------- launcher ----------------
extern "C" void kernel_launch(void* const* d_in, const int* in_sizes, int n_in,
                              void* d_out, int out_size, void* d_ws, size_t ws_size,
                              hipStream_t stream) {
  (void)in_sizes; (void)n_in; (void)out_size; (void)d_ws; (void)ws_size;
  const float* embs = (const float*)d_in[0];  // last_embs [B][N][D]
  const float* neib = (const float*)d_in[1];  // neibors   [B][N][N]
  float* out        = (float*)d_out;          // [B][N][D]
  dim3 grid(kN / BM, kB);
  Aggregators_87170656239792_kernel<<<grid, 256, 0, stream>>>(embs, neib, out);
}
